// ProtoNet_55843164783336
// MI455X (gfx1250) — matrix operand order repaired and checked by reading, not compiled, not run on hardware
//
#include <hip/hip_runtime.h>

// ProtoNet NES-distance on MI455X (gfx1250, wave32, WMMA).
//
// Shapes: N=64 S=512 E=768 P=64 K=4 C=4, D=K*E=3072.
// Branch 0: dil=1, H=509, protos [0,32). Branch 1: dil=2, H=506, protos [32,64).
//
// The reference's reshape(N,H,K*E) of the (N,K,H,E) gather is a raw reshape:
// wnd[n,h,:] == 3072 consecutive floats of concat_k(embedding[n, k*d : k*d+H, :]).
// Window stride == window length, so the unfold is a plain (H x 3072) matrix whose
// 768-float column chunks are contiguous, 16B-aligned slices of embedding.
//
// Roofline: cross-GEMM = 12.8 GFLOP vs 100 MB traffic (AI ~128 flop/B) -> compute
// bound in f32 WMMA (16x16x4). f16 WMMA (16x16x32, f32 accumulate) is 8x denser,
// balancing against the 23.3 TB/s HBM time (~4-5us). Sums/variances/epilogue in f32.

#define NN 64
#define SS 512
#define EE 768
#define PP 64
#define KW 4
#define CC 4
#define DD (KW * EE)          // 3072
#define EPSF 1e-8f

typedef __attribute__((ext_vector_type(16))) _Float16 v16h;
typedef __attribute__((ext_vector_type(8)))  _Float16 v8h;
typedef __attribute__((ext_vector_type(8)))  float    v8f;

// Order-preserving float -> uint map (for global_atomic_min_u32).
__device__ __forceinline__ unsigned orderKey(float x) {
    int b = __float_as_int(x);
    unsigned m = (b < 0) ? 0xFFFFFFFFu : 0x80000000u;
    return ((unsigned)b) ^ m;
}
__device__ __forceinline__ float orderDecode(unsigned k) {
    unsigned b = (k & 0x80000000u) ? (k ^ 0x80000000u) : ~k;
    return __uint_as_float(b);
}

// ---------------------------------------------------------------------------
// Kernel 1: per-prototype f16 conversion + stats, and min-buffer re-init.
// grid=64 (one block per prototype), block=256.
__global__ void proto_prep(const float* __restrict__ proto,
                           _Float16* __restrict__ protoF16,
                           float* __restrict__ sp, float* __restrict__ vp,
                           unsigned* __restrict__ minbuf) {
    int p = blockIdx.x;
    int t = threadIdx.x;
    if (t < PP) minbuf[p * PP + t] = 0xFFFFFFFFu;   // +inf key, re-init every launch

    const float* row = proto + (size_t)p * DD;      // pflat row p is contiguous
    float s = 0.f, s2 = 0.f;
    for (int i = t; i < DD; i += 256) {
        float v = row[i];
        protoF16[(size_t)p * DD + i] = (_Float16)v;
        s += v; s2 += v * v;
    }
    __shared__ float rs[256], rq[256];
    rs[t] = s; rq[t] = s2;
    __syncthreads();
    for (int o = 128; o > 0; o >>= 1) {
        if (t < o) { rs[t] += rs[t + o]; rq[t] += rq[t + o]; }
        __syncthreads();
    }
    if (t == 0) {
        float S = rs[0], Q = rq[0];
        sp[p] = S;
        vp[p] = (Q - S * S / (float)DD) / (float)(DD - 1);
    }
}

// ---------------------------------------------------------------------------
// Kernel 2: main WMMA kernel.
// grid = 2 branches * 64 n * 32 h-tiles = 4096 WGs; block = 64 (2 waves).
// Wave w handles proto tile [branch*32 + w*16, +16).
// Dynamic LDS: A tile 16 rows x 3072 f16 = 96 KB (3 WGs / 320KB WGP).
__global__ void __launch_bounds__(64)
nes_wmma(const float* __restrict__ emb,
         const float* __restrict__ mask,
         const _Float16* __restrict__ protoF16,
         const float* __restrict__ sp, const float* __restrict__ vp,
         unsigned* __restrict__ minbuf) {
    extern __shared__ _Float16 smemA[];             // [16][3072]
    __shared__ float sSx[16], sSxx[16], sVx[16];

    int bid    = blockIdx.x;
    int branch = bid >> 11;                          // / 2048
    int rem    = bid & 2047;
    int n      = rem >> 5;                           // / 32
    int ht     = rem & 31;
    int dil     = branch ? 2 : 1;
    int H       = SS - dil * (KW - 1);               // 509 / 506
    int overlap = dil * (KW - 1);
    int HE      = H * EE;                            // block length in concat buffer
    int tid     = threadIdx.x;

    if (tid < 16) { sSx[tid] = 0.f; sSxx[tid] = 0.f; }
    __syncthreads();

    // ---- Stage A tile: 16 rows x 3072 f32 -> f16 into LDS, + row sum/sumsq.
    // 64 chunks = 16 rows x 4 contiguous 768-float pieces; each chunk is a
    // contiguous, 16B-aligned slice of embedding (block boundaries are
    // multiples of 768). All 64 threads load one chunk together (coalesced).
    const float* embN = emb + (size_t)n * (SS * EE);
    for (int ch = 0; ch < 64; ++ch) {
        int r = ch >> 2, g = ch & 3;
        int h  = ht * 16 + r;
        int hl = (h < H) ? h : (H - 1);              // clamp: dup row for h>=H (skipped later)
        int i  = hl * DD + g * EE;                   // flat offset in concat buffer
        int k  = i / HE;                             // which of the 4 slices
        int o  = i - k * HE;                         // offset within slice
        const float4* src = (const float4*)(embN + (size_t)k * dil * EE + o);
        _Float16* dst = smemA + r * DD + g * EE;
        float s = 0.f, s2 = 0.f;
        #pragma unroll
        for (int q = 0; q < 3; ++q) {                // 192 float4 / 64 threads
            int v4 = tid + 64 * q;
            float4 x = src[v4];
            int fo = v4 * 4;
            dst[fo + 0] = (_Float16)x.x; dst[fo + 1] = (_Float16)x.y;
            dst[fo + 2] = (_Float16)x.z; dst[fo + 3] = (_Float16)x.w;
            s  += x.x + x.y + x.z + x.w;
            s2 += x.x * x.x + x.y * x.y + x.z * x.z + x.w * x.w;
        }
        atomicAdd(&sSx[r], s);                       // ds_add_f32
        atomicAdd(&sSxx[r], s2);
    }
    __syncthreads();
    if (tid < 16) {
        float sx = sSx[tid];
        sVx[tid] = (sSxx[tid] - sx * sx / (float)DD) / (float)(DD - 1);
    }
    __syncthreads();

    // ---- WMMA: C[16h x 16p] += A[16 x 32] * B[32 x 16], 96 K-steps over D=3072.
    // A layout (16-bit A 16x32): lane L<16 -> M=L, K {0..15};
    // lane>=16 -> M=L-16, K {16..31} (matches B). Two 16B LDS chunks per step.
    // B layout (16-bit B 32x16): lane = N, halves K = hiHalf*16 + 0..15 contiguous.
    int wv     = tid >> 5;
    int lane   = tid & 31;
    int laneN  = lane & 15;
    int hiHalf = lane >> 4;
    int pTile  = branch * 32 + wv * 16;
    int p      = pTile + laneN;

    const _Float16* ap = smemA + laneN * DD + hiHalf * 16;
    const _Float16* bp = protoF16 + (size_t)p * DD + hiHalf * 16;

    v8f c = {0.f, 0.f, 0.f, 0.f, 0.f, 0.f, 0.f, 0.f};
    #pragma unroll 4
    for (int kk = 0; kk < 96; ++kk) {
        v8h a0 = *(const v8h*)(ap + kk * 32);        // K base+{0..7}   (hi: +16)
        v8h a1 = *(const v8h*)(ap + kk * 32 + 8);    // K base+{8..15}  (hi: +24)
        v16h a;
        #pragma unroll
        for (int i = 0; i < 8; ++i) { a[i] = a0[i]; a[i + 8] = a1[i]; }
        v16h b = *(const v16h*)(bp + kk * 32);       // contiguous 16 halves
        c = __builtin_amdgcn_wmma_f32_16x16x32_f16(
                /*neg_a=*/false, a, /*neg_b=*/false, b,
                /*c_mod=*/(short)0, c, /*reuse_a=*/false, /*reuse_b=*/false);
    }

    // ---- Epilogue: NES distance, mask, masked min over h.
    // C layout: lane L -> N=L%16; VGPR r -> M = r + 8*(L>=16).
    float spv = sp[p], vpv = vp[p];
    const float invD   = 1.f / (float)DD;
    const float invDm1 = 1.f / (float)(DD - 1);
    float mn = 3.0e38f;
    #pragma unroll
    for (int r = 0; r < 8; ++r) {
        int M = r + hiHalf * 8;
        int h = ht * 16 + M;
        if (h < H) {
            float sx   = sSx[M], vx = sVx[M];
            float cov  = (c[r] - sx * spv * invD) * invDm1;
            float vsum = vx + vpv;
            float ned2 = 0.5f * (vsum - 2.f * cov) / (vsum + EPSF);
            float dist = sqrtf(fmaxf(ned2, 0.f)) - 1.f;
            float mval = (h == 0) ? 1.f : mask[n * SS + overlap + h];  // m.at[:,overlap]=1
            mn = fminf(mn, dist * mval);
        }
    }
    mn = fminf(mn, __shfl_xor(mn, 16));              // fold M 0..7 with 8..15 (same N)
    if (lane < 16) atomicMin(minbuf + n * PP + p, orderKey(mn));
}

// ---------------------------------------------------------------------------
// Kernel 3: decode mins -> proto_dist, tiny classifier GEMV. 1 block x 256.
__global__ void finalize_k(const unsigned* __restrict__ minbuf,
                           const float* __restrict__ fcw,
                           float* __restrict__ out) {
    __shared__ float pd[NN * PP];
    int t = threadIdx.x;
    for (int i = t; i < NN * PP; i += 256) {
        float v = orderDecode(minbuf[i]);
        pd[i] = v;
        out[NN * CC + i] = v;                        // proto_dist after class_out
    }
    __syncthreads();
    int nn = t >> 2, cc = t & 3;                     // 256 == 64*4 exactly
    const float* w   = fcw + cc * PP;                // fc_w is (C,P)
    const float* row = pd + nn * PP;
    float s = 0.f;
    #pragma unroll 8
    for (int q = 0; q < PP; ++q) s += row[q] * w[q];
    out[nn * CC + cc] = s;                           // class_out = pd @ fc_w.T
}

// ---------------------------------------------------------------------------
extern "C" void kernel_launch(void* const* d_in, const int* in_sizes, int n_in,
                              void* d_out, int out_size, void* d_ws, size_t ws_size,
                              hipStream_t stream) {
    const float* emb   = (const float*)d_in[0];   // (64,512,768)
    const float* mask  = (const float*)d_in[1];   // (64,512)
    const float* proto = (const float*)d_in[2];   // (1,64,768,4) == pflat (64,3072)
    const float* fcw   = (const float*)d_in[3];   // (4,64)
    float* out = (float*)d_out;                   // class_out(256) ++ proto_dist(4096)

    char* ws = (char*)d_ws;
    _Float16* protoF16 = (_Float16*)ws;                        // 393216 B
    float*    sp       = (float*)(ws + 393216);                // 256 B
    float*    vp       = (float*)(ws + 393216 + 256);          // 256 B
    unsigned* minbuf   = (unsigned*)(ws + 393216 + 512);       // 16384 B

    proto_prep<<<PP, 256, 0, stream>>>(proto, protoF16, sp, vp, minbuf);
    nes_wmma<<<2 * NN * 32, 64, (size_t)16 * DD * sizeof(_Float16), stream>>>(
        emb, mask, protoF16, sp, vp, minbuf);
    finalize_k<<<1, 256, 0, stream>>>(minbuf, fcw, out);
}
